// DeformableSiameseAttention_28097676050805
// MI455X (gfx1250) — compile-verified, matrix operand order checked by reading, not run on hardware
//
#include <hip/hip_runtime.h>
#include <hip/hip_bf16.h>
#include <cmath>

typedef __attribute__((ext_vector_type(16))) _Float16 v16h;
typedef __attribute__((ext_vector_type(8)))  _Float16 v8h;
typedef __attribute__((ext_vector_type(8)))  float    v8f;

// Block tile: 64(m) x 128(n), K-step 32. 256 threads = 8 waves, 2(m) x 4(n);
// each wave computes a 32x32 patch = four 16x16 WMMA accumulators, so each
// A/B fragment fetched from LDS is reused by two WMMAs (2 ds_load_b128/WMMA).
#define TBM 64
#define TBN 128
#define TBK 32
#define TKP 40   // padded row pitch in halves: 80B, 16B-aligned rows

// ---------------------------------------------------------------------------
// Generic batched GEMM: D[b] (MxN, row-major, ld=N) = A[b] * B[b] (+bias)(+acc)
// A element (m,k) at A + b*a_bs + m*a_rs + k*a_cs   (strides in elements)
// B element (k,n) at B + b*b_bs + k*b_rs + n*b_cs
// f32/f16 inputs staged to LDS as f16 (branchless clamped loads, double
// buffered), accumulated in f32 via v_wmma_f32_16x16x32_f16.
// ---------------------------------------------------------------------------
template <typename TB>
__global__ __launch_bounds__(256)
void wmma_gemm_kernel(const float* __restrict__ A, long a_bs, long a_rs, long a_cs,
                      const TB*    __restrict__ B, long b_bs, long b_rs, long b_cs,
                      float*       __restrict__ D, long d_bs,
                      int M, int N, int K,
                      const float* __restrict__ bias,
                      const float* __restrict__ scale,
                      int accumulate)
{
    __shared__ alignas(16) _Float16 As[2][TBM][TKP];        // A tile, row-major
    __shared__ alignas(16) _Float16 Bs[2][TBN][TKP];        // B tile, TRANSPOSED: [n][k]

    const int b  = blockIdx.z;
    const float* Ab = A + (long)b * a_bs;
    const TB*    Bb = B + (long)b * b_bs;
    float*       Db = D + (long)b * d_bs;

    const int m0   = blockIdx.x * TBM;
    const int n0   = blockIdx.y * TBN;
    const int tid  = threadIdx.x;
    const int lane = tid & 31;
    const int wave = tid >> 5;
    const int wm   = wave >> 2;   // 0..1
    const int wn   = wave & 3;    // 0..3

    // Branchless clamped staging: always-legal address, zero via mask multiply.
    auto stage = [&](int k0, int p) {
#pragma unroll
        for (int it = 0; it < (TBM * TBK) / 256; ++it) {     // 8 iters
            int i  = tid + it * 256;
            int r  = i >> 5, c = i & 31;                     // TBK == 32
            int gm = m0 + r, gk = k0 + c;
            int ok = (gm < M) & (gk < K);
            long idx = (long)(gm < M ? gm : M - 1) * a_rs
                     + (long)(gk < K ? gk : K - 1) * a_cs;
            float v = Ab[idx] * (float)ok;
            As[p][r][c] = (_Float16)v;
        }
#pragma unroll
        for (int it = 0; it < (TBN * TBK) / 256; ++it) {     // 16 iters
            int i  = tid + it * 256;
            int r  = i >> 7, c = i & 127;                    // r = k idx, c = n idx
            int gk = k0 + r, gn = n0 + c;
            int ok = (gk < K) & (gn < N);
            long idx = (long)(gk < K ? gk : K - 1) * b_rs
                     + (long)(gn < N ? gn : N - 1) * b_cs;
            float v = (float)Bb[idx] * (float)ok;
            Bs[p][c][r] = (_Float16)v;                        // transposed store
        }
    };

    v8f acc00 = {};   // (row tile 0, col tile 0)
    v8f acc01 = {};   // (row tile 0, col tile 1)
    v8f acc10 = {};   // (row tile 1, col tile 0)
    v8f acc11 = {};   // (row tile 1, col tile 1)

    const int row0 = wm * 32 + (lane & 15);       // A rows within tile
    const int row1 = row0 + 16;
    const int col0 = wn * 32 + (lane & 15);       // B cols within tile
    const int col1 = col0 + 16;
    const int kb   = (lane < 16) ? 0 : 8;         // per-lane K base (16x16x32 layout)

    int p = 0;
    stage(0, 0);
    __syncthreads();

    for (int k0 = 0; k0 < K; k0 += TBK) {
        if (k0 + TBK < K) stage(k0 + TBK, p ^ 1);  // overlap next-tile staging

        // Fragments: 8 contiguous halves at 16B-aligned addresses -> ds_load_b128
        const _Float16* Ar0 = &As[p][row0][0];
        const _Float16* Ar1 = &As[p][row1][0];
        const _Float16* Br0 = &Bs[p][col0][0];
        const _Float16* Br1 = &Bs[p][col1][0];
        v8h a00 = *(const v8h*)(Ar0 + kb);
        v8h a01 = *(const v8h*)(Ar0 + kb + 16);
        v8h a10 = *(const v8h*)(Ar1 + kb);
        v8h a11 = *(const v8h*)(Ar1 + kb + 16);
        v8h b00 = *(const v8h*)(Br0 + kb);
        v8h b01 = *(const v8h*)(Br0 + kb + 16);
        v8h b10 = *(const v8h*)(Br1 + kb);
        v8h b11 = *(const v8h*)(Br1 + kb + 16);
        v16h af0 = __builtin_shufflevector(a00, a01, 0,1,2,3,4,5,6,7,8,9,10,11,12,13,14,15);
        v16h af1 = __builtin_shufflevector(a10, a11, 0,1,2,3,4,5,6,7,8,9,10,11,12,13,14,15);
        v16h bf0 = __builtin_shufflevector(b00, b01, 0,1,2,3,4,5,6,7,8,9,10,11,12,13,14,15);
        v16h bf1 = __builtin_shufflevector(b10, b11, 0,1,2,3,4,5,6,7,8,9,10,11,12,13,14,15);

        acc00 = __builtin_amdgcn_wmma_f32_16x16x32_f16(false, af0, false, bf0,
                                                       (short)0, acc00, false, false);
        acc01 = __builtin_amdgcn_wmma_f32_16x16x32_f16(false, af0, false, bf1,
                                                       (short)0, acc01, false, false);
        acc10 = __builtin_amdgcn_wmma_f32_16x16x32_f16(false, af1, false, bf0,
                                                       (short)0, acc10, false, false);
        acc11 = __builtin_amdgcn_wmma_f32_16x16x32_f16(false, af1, false, bf1,
                                                       (short)0, acc11, false, false);
        __syncthreads();
        p ^= 1;
    }

    // Epilogue. C/D layout: VGPR i -> M = i (lanes 0-15) or 8+i (lanes 16-31); N = lane&15.
    const float sc = scale ? *scale : 1.0f;
    const int gn0 = n0 + wn * 32 + (lane & 15);
    const int gn1 = gn0 + 16;
    const int mb0 = m0 + wm * 32 + ((lane >> 4) << 3);
    const int mb1 = mb0 + 16;
#pragma unroll
    for (int i = 0; i < 8; ++i) {
        int gmA = mb0 + i;
        int gmB = mb1 + i;
        if (gmA < M) {
            float bv = bias ? bias[gmA] : 0.0f;
            if (gn0 < N) {
                float v = acc00[i] * sc + bv;
                long off = (long)gmA * N + gn0;
                Db[off] = accumulate ? (Db[off] + v) : v;
            }
            if (gn1 < N) {
                float v = acc01[i] * sc + bv;
                long off = (long)gmA * N + gn1;
                Db[off] = accumulate ? (Db[off] + v) : v;
            }
        }
        if (gmB < M) {
            float bv = bias ? bias[gmB] : 0.0f;
            if (gn0 < N) {
                float v = acc10[i] * sc + bv;
                long off = (long)gmB * N + gn0;
                Db[off] = accumulate ? (Db[off] + v) : v;
            }
            if (gn1 < N) {
                float v = acc11[i] * sc + bv;
                long off = (long)gmB * N + gn1;
                Db[off] = accumulate ? (Db[off] + v) : v;
            }
        }
    }
}

// ---------------------------------------------------------------------------
// In-place row softmax. One block per row. negate=1 computes softmax(-x),
// equivalent to the reference softmax(rowmax - x) (shift invariant).
// ---------------------------------------------------------------------------
__global__ __launch_bounds__(256)
void softmax_rows_kernel(float* __restrict__ E, int N, int negate)
{
    __shared__ float red[256];
    float* row = E + (long)blockIdx.x * N;
    const int tid = threadIdx.x;

    float mx = -3.4e38f;
    for (int i = tid; i < N; i += 256) {
        float v = negate ? -row[i] : row[i];
        mx = fmaxf(mx, v);
    }
    red[tid] = mx; __syncthreads();
    for (int s = 128; s > 0; s >>= 1) { if (tid < s) red[tid] = fmaxf(red[tid], red[tid + s]); __syncthreads(); }
    mx = red[0]; __syncthreads();

    float sum = 0.0f;
    for (int i = tid; i < N; i += 256) {
        float v = __expf((negate ? -row[i] : row[i]) - mx);
        row[i] = v;
        sum += v;
    }
    red[tid] = sum; __syncthreads();
    for (int s = 128; s > 0; s >>= 1) { if (tid < s) red[tid] += red[tid + s]; __syncthreads(); }
    const float inv = 1.0f / red[0];
    for (int i = tid; i < N; i += 256) row[i] *= inv;
}

__global__ __launch_bounds__(256)
void copy_kernel(float* __restrict__ dst, const float* __restrict__ src, long n)
{
    long i = (long)blockIdx.x * blockDim.x + threadIdx.x;
    if (i < n) dst[i] = src[i];
}

// ---------------------------------------------------------------------------
// Direct 3x3 conv (pad=1): om[b, oc, h, w], 27 output channels of offsets/mask.
// Per-oc 2304-float weight row cached in LDS. grid=(ceil(N/128), 27, B)
// ---------------------------------------------------------------------------
__global__ __launch_bounds__(128)
void offset_conv_kernel(const float* __restrict__ x, const float* __restrict__ w,
                        const float* __restrict__ bias, float* __restrict__ om,
                        int H, int W, int C)
{
    __shared__ float ws[2304];
    const int oc = blockIdx.y, b = blockIdx.z;
    const int N = H * W;
    for (int i = threadIdx.x; i < C * 9; i += 128) ws[i] = w[(long)oc * C * 9 + i];
    __syncthreads();

    const int n = blockIdx.x * 128 + threadIdx.x;
    if (n >= N) return;
    const int h = n / W, wp = n % W;
    const float* xb = x + (long)b * C * N;

    float sum = bias[oc];
    for (int c = 0; c < C; ++c) {
        const float* xc = xb + (long)c * N;
        const float* wc = ws + c * 9;
#pragma unroll
        for (int ky = 0; ky < 3; ++ky) {
            int yy = h + ky - 1;
            if (yy < 0 || yy >= H) continue;
#pragma unroll
            for (int kx = 0; kx < 3; ++kx) {
                int xx = wp + kx - 1;
                if (xx < 0 || xx >= W) continue;
                sum += xc[yy * W + xx] * wc[ky * 3 + kx];
            }
        }
    }
    om[((long)b * 27 + oc) * N + n] = sum;
}

// ---------------------------------------------------------------------------
// Deformable bilinear gather -> f16 "deformable im2col" val[(c*9+k), n].
// One thread per (tap k, pixel n); offsets/mask computed once, 256-channel loop.
// grid=(ceil(N/128), 9, B)
// ---------------------------------------------------------------------------
__global__ __launch_bounds__(128)
void dcn_gather_kernel(const float* __restrict__ x, const float* __restrict__ om,
                       _Float16* __restrict__ val, int H, int W, int C)
{
    const int k = blockIdx.y, b = blockIdx.z;
    const int N = H * W;
    const int n = blockIdx.x * 128 + threadIdx.x;
    if (n >= N) return;
    const int h = n / W, wp = n % W;

    const float* omb = om + (long)b * 27 * N;
    const float oy = omb[(long)k * N + n];
    const float ox = omb[(long)(9 + k) * N + n];
    float msk = omb[(long)(18 + k) * N + n];
    msk = 1.0f / (1.0f + __expf(-msk));

    const int ky = k / 3, kx = k % 3;
    const float py = oy + (float)h + (float)(ky - 1);
    const float px = ox + (float)wp + (float)(kx - 1);
    const float y0f = floorf(py), x0f = floorf(px);
    const float wy = py - y0f, wx = px - x0f;
    const int y0 = (int)y0f, x0 = (int)x0f;
    const int y1 = y0 + 1, x1 = x0 + 1;

    const float w00 = (1.0f - wy) * (1.0f - wx) * ((y0 >= 0 && y0 < H && x0 >= 0 && x0 < W) ? 1.0f : 0.0f);
    const float w01 = (1.0f - wy) * wx          * ((y0 >= 0 && y0 < H && x1 >= 0 && x1 < W) ? 1.0f : 0.0f);
    const float w10 = wy * (1.0f - wx)          * ((y1 >= 0 && y1 < H && x0 >= 0 && x0 < W) ? 1.0f : 0.0f);
    const float w11 = wy * wx                   * ((y1 >= 0 && y1 < H && x1 >= 0 && x1 < W) ? 1.0f : 0.0f);

    const int y0c = min(max(y0, 0), H - 1), y1c = min(max(y1, 0), H - 1);
    const int x0c = min(max(x0, 0), W - 1), x1c = min(max(x1, 0), W - 1);
    const int i00 = y0c * W + x0c, i01 = y0c * W + x1c;
    const int i10 = y1c * W + x0c, i11 = y1c * W + x1c;

    const float* xb = x + (long)b * C * N;
    _Float16* vb = val + (long)b * C * 9 * N;
    for (int c = 0; c < C; ++c) {
        const float* xc = xb + (long)c * N;
        float v = w00 * xc[i00] + w01 * xc[i01] + w10 * xc[i10] + w11 * xc[i11];
        vb[(long)(c * 9 + k) * N + n] = (_Float16)(v * msk);
    }
}

// ---------------------------------------------------------------------------
extern "C" void kernel_launch(void* const* d_in, const int* in_sizes, int n_in,
                              void* d_out, int out_size, void* d_ws, size_t ws_size,
                              hipStream_t stream)
{
    (void)in_sizes; (void)n_in; (void)out_size; (void)ws_size;

    const int Bn = 16, C = 256, QK = 32;
    const int Ht = 15, Hs = 31;
    const int Nt = Ht * Ht;   // 225
    const int Ns = Hs * Hs;   // 961

    const float* t_in    = (const float*)d_in[0];
    const float* s_in    = (const float*)d_in[1];
    const float* t_qw    = (const float*)d_in[2];
    const float* t_qb    = (const float*)d_in[3];
    const float* t_kw    = (const float*)d_in[4];
    const float* t_kb    = (const float*)d_in[5];
    const float* t_vw    = (const float*)d_in[6];
    const float* t_vb    = (const float*)d_in[7];
    const float* t_g     = (const float*)d_in[8];
    const float* s_qw    = (const float*)d_in[9];
    const float* s_qb    = (const float*)d_in[10];
    const float* s_kw    = (const float*)d_in[11];
    const float* s_kb    = (const float*)d_in[12];
    const float* s_vw    = (const float*)d_in[13];
    const float* s_vb    = (const float*)d_in[14];
    const float* s_g     = (const float*)d_in[15];
    const float* w_t_ch  = (const float*)d_in[16];
    const float* w_s_ch  = (const float*)d_in[17];
    const float* w_s2t   = (const float*)d_in[18];
    const float* w_t2s   = (const float*)d_in[19];
    const float* t_off_w = (const float*)d_in[20];
    const float* t_off_b = (const float*)d_in[21];
    const float* t_dcn_w = (const float*)d_in[22];
    const float* t_dcn_b = (const float*)d_in[23];
    const float* s_off_w = (const float*)d_in[24];
    const float* s_off_b = (const float*)d_in[25];
    const float* s_dcn_w = (const float*)d_in[26];
    const float* s_dcn_b = (const float*)d_in[27];

    // ---- workspace layout (persistent sums, then stage scratch aliased) ----
    char* ws = (char*)d_ws;
    size_t off = 0;
    auto allocf = [&](size_t n) -> float* {
        float* p = (float*)(ws + off); off += n * sizeof(float); return p;
    };
    float* tsum = allocf((size_t)Bn * C * Nt);
    float* ssum = allocf((size_t)Bn * C * Ns);
    const size_t stage_off = off;

    // stage 1: attention scratch
    float* qt = allocf((size_t)Bn * QK * Nt);
    float* kt = allocf((size_t)Bn * QK * Nt);
    float* vt = allocf((size_t)Bn * C  * Nt);
    float* Et = allocf((size_t)Bn * Nt * Nt);
    float* qs = allocf((size_t)Bn * QK * Ns);
    float* ks = allocf((size_t)Bn * QK * Ns);
    float* vs = allocf((size_t)Bn * C  * Ns);
    float* Es = allocf((size_t)Bn * Ns * Ns);
    float* tA = allocf((size_t)Bn * C * C);
    float* sA = allocf((size_t)Bn * C * C);

    // stage 2: DCN scratch, aliases stage 1 (stage-1 values dead by then)
    off = stage_off;
    float* omt = allocf((size_t)Bn * 27 * Nt);
    float* oms = allocf((size_t)Bn * 27 * Ns);
    _Float16* valt = (_Float16*)(ws + off); off += (size_t)Bn * C * 9 * Nt * sizeof(_Float16);
    _Float16* vals = (_Float16*)(ws + off); off += (size_t)Bn * C * 9 * Ns * sizeof(_Float16);

    const dim3 blk(256);
    auto gemmF = [&](const float* A, long abs_, long ars, long acs,
                     const float* Bm, long bbs, long brs, long bcs,
                     float* D, long dbs, int M, int N, int K,
                     const float* bias, const float* sc, int accum) {
        dim3 grid((M + TBM - 1) / TBM, (N + TBN - 1) / TBN, Bn);
        wmma_gemm_kernel<float><<<grid, blk, 0, stream>>>(
            A, abs_, ars, acs, Bm, bbs, brs, bcs, D, dbs, M, N, K, bias, sc, accum);
    };
    auto gemmH = [&](const float* A, long abs_, long ars, long acs,
                     const _Float16* Bm, long bbs, long brs, long bcs,
                     float* D, long dbs, int M, int N, int K,
                     const float* bias, const float* sc, int accum) {
        dim3 grid((M + TBM - 1) / TBM, (N + TBN - 1) / TBN, Bn);
        wmma_gemm_kernel<_Float16><<<grid, blk, 0, stream>>>(
            A, abs_, ars, acs, Bm, bbs, brs, bcs, D, dbs, M, N, K, bias, sc, accum);
    };

    // ---- 1) q/k/v 1x1 convs:  W(OxC) * X(CxN) + bias ----
    gemmF(t_qw, 0, C, 1, t_in, (long)C * Nt, Nt, 1, qt, (long)QK * Nt, QK, Nt, C, t_qb, nullptr, 0);
    gemmF(t_kw, 0, C, 1, t_in, (long)C * Nt, Nt, 1, kt, (long)QK * Nt, QK, Nt, C, t_kb, nullptr, 0);
    gemmF(t_vw, 0, C, 1, t_in, (long)C * Nt, Nt, 1, vt, (long)C  * Nt, C,  Nt, C, t_vb, nullptr, 0);
    gemmF(s_qw, 0, C, 1, s_in, (long)C * Ns, Ns, 1, qs, (long)QK * Ns, QK, Ns, C, s_qb, nullptr, 0);
    gemmF(s_kw, 0, C, 1, s_in, (long)C * Ns, Ns, 1, ks, (long)QK * Ns, QK, Ns, C, s_kb, nullptr, 0);
    gemmF(s_vw, 0, C, 1, s_in, (long)C * Ns, Ns, 1, vs, (long)C  * Ns, C,  Ns, C, s_vb, nullptr, 0);

    // ---- 2) energy E[n,m] = sum_q q[q,n]*k[q,m]  (A = q^T) ----
    gemmF(qt, (long)QK * Nt, 1, Nt, kt, (long)QK * Nt, Nt, 1, Et, (long)Nt * Nt, Nt, Nt, QK, nullptr, nullptr, 0);
    gemmF(qs, (long)QK * Ns, 1, Ns, ks, (long)QK * Ns, Ns, 1, Es, (long)Ns * Ns, Ns, Ns, QK, nullptr, nullptr, 0);

    // ---- 3) spatial softmax over m (rows) ----
    softmax_rows_kernel<<<Bn * Nt, blk, 0, stream>>>(Et, Nt, 0);
    softmax_rows_kernel<<<Bn * Ns, blk, 0, stream>>>(Es, Ns, 0);

    // ---- 4) channel energy e = f f^T, then softmax(max-e) == softmax(-e) ----
    gemmF(t_in, (long)C * Nt, Nt, 1, t_in, (long)C * Nt, 1, Nt, tA, (long)C * C, C, C, Nt, nullptr, nullptr, 0);
    gemmF(s_in, (long)C * Ns, Ns, 1, s_in, (long)C * Ns, 1, Ns, sA, (long)C * C, C, C, Ns, nullptr, nullptr, 0);
    softmax_rows_kernel<<<Bn * C, blk, 0, stream>>>(tA, C, 1);
    softmax_rows_kernel<<<Bn * C, blk, 0, stream>>>(sA, C, 1);

    // ---- 5) residual init ----
    {
        long nt = (long)Bn * C * Nt, ns = (long)Bn * C * Ns;
        copy_kernel<<<(unsigned)((nt + 255) / 256), blk, 0, stream>>>(tsum, t_in, nt);
        copy_kernel<<<(unsigned)((ns + 255) / 256), blk, 0, stream>>>(ssum, s_in, ns);
    }

    // ---- 6) spatial apply: sum += g * v(CxN) * A^T  (B[m,n] = E[n,m]) ----
    gemmF(vt, (long)C * Nt, Nt, 1, Et, (long)Nt * Nt, 1, Nt, tsum, (long)C * Nt, C, Nt, Nt, nullptr, t_g, 1);
    gemmF(vs, (long)C * Ns, Ns, 1, Es, (long)Ns * Ns, 1, Ns, ssum, (long)C * Ns, C, Ns, Ns, nullptr, s_g, 1);

    // ---- 7) channel uses: sum += w * A(CxC) * x(CxN) ----
    gemmF(tA, (long)C * C, C, 1, t_in, (long)C * Nt, Nt, 1, tsum, (long)C * Nt, C, Nt, C, nullptr, w_t_ch, 1);
    gemmF(sA, (long)C * C, C, 1, t_in, (long)C * Nt, Nt, 1, tsum, (long)C * Nt, C, Nt, C, nullptr, w_s2t,  1);
    gemmF(sA, (long)C * C, C, 1, s_in, (long)C * Ns, Ns, 1, ssum, (long)C * Ns, C, Ns, C, nullptr, w_s_ch, 1);
    gemmF(tA, (long)C * C, C, 1, s_in, (long)C * Ns, Ns, 1, ssum, (long)C * Ns, C, Ns, C, nullptr, w_t2s,  1);

    // ---- 8) offset/mask 3x3 conv ----
    offset_conv_kernel<<<dim3((Nt + 127) / 128, 27, Bn), dim3(128), 0, stream>>>(tsum, t_off_w, t_off_b, omt, Ht, Ht, C);
    offset_conv_kernel<<<dim3((Ns + 127) / 128, 27, Bn), dim3(128), 0, stream>>>(ssum, s_off_w, s_off_b, oms, Hs, Hs, C);

    // ---- 9) deformable bilinear gather -> f16 im2col ----
    dcn_gather_kernel<<<dim3((Nt + 127) / 128, 9, Bn), dim3(128), 0, stream>>>(tsum, omt, valt, Ht, Ht, C);
    dcn_gather_kernel<<<dim3((Ns + 127) / 128, 9, Bn), dim3(128), 0, stream>>>(ssum, oms, vals, Hs, Hs, C);

    // ---- 10) DCN main GEMM: out = W(256x2304) * val(2304xN) + bias ----
    float* out_t = (float*)d_out;
    float* out_s = out_t + (size_t)Bn * C * Nt;
    gemmH(t_dcn_w, 0, 9 * C, 1, valt, (long)C * 9 * Nt, Nt, 1, out_t, (long)C * Nt, C, Nt, 9 * C, t_dcn_b, nullptr, 0);
    gemmH(s_dcn_w, 0, 9 * C, 1, vals, (long)C * 9 * Ns, Ns, 1, out_s, (long)C * Ns, C, Ns, 9 * C, s_dcn_b, nullptr, 0);
}
